// Task_Aware_Delta_Net_3358664426168
// MI455X (gfx1250) — matrix operand order repaired and checked by reading, not compiled, not run on hardware
//
#include <hip/hip_runtime.h>
#include <cstdint>

#define HIDDEN   2048
#define NUM_HEADS 6
#define HEAD_DIM 256
#define KEY_DIM  (NUM_HEADS * HEAD_DIM)   // 1536
#define CONV_K   4

typedef __attribute__((ext_vector_type(16))) __bf16 v16bf;
typedef __attribute__((ext_vector_type(8)))  float  v8f;

// ---------- helpers ----------
__device__ __forceinline__ unsigned short f32_to_bf16(float f) {
  union { float f; uint32_t u; } v; v.f = f;
  uint32_t u = v.u;
  uint32_t rnd = 0x7FFFu + ((u >> 16) & 1u);   // round-to-nearest-even
  return (unsigned short)((u + rnd) >> 16);
}
__device__ __forceinline__ __bf16 us_to_bf(unsigned short u) {
  union { unsigned short u; __bf16 b; } c; c.u = u; return c.b;
}
__device__ __forceinline__ float silu(float x) {
  return x / (1.f + __expf(-x));
}

// async global->LDS copy (CDNA5 VGLOBAL async path, ASYNCcnt-tracked).
// Copies 16B/lane at (lds + 0 / +16) from (gaddr + 0 / +16); IOFFSET is added
// to both LDS and global addresses per the ISA pseudocode.
__device__ __forceinline__ void async_copy_32B(uint32_t lds_addr, const void* gaddr) {
  asm volatile("global_load_async_to_lds_b128 %0, %1, off\n\t"
               "global_load_async_to_lds_b128 %0, %1, off offset:16"
               :: "v"(lds_addr), "v"(gaddr) : "memory");
}
__device__ __forceinline__ void wait_async0() {
  asm volatile("s_wait_asynccnt 0x0" ::: "memory");
}

// ---------- fp32 -> bf16 convert ----------
__global__ void cvt_bf16_kernel(const float* __restrict__ src,
                                unsigned short* __restrict__ dst, int n) {
  int i = blockIdx.x * 256 + threadIdx.x;
  if (i < n) dst[i] = f32_to_bf16(src[i]);
}

// ---------- bf16 WMMA GEMM: C[M,N] = A[M,K] * W[N,K]^T ----------
// 256 threads (8 waves). Block tile 128x128, BK=32, double-buffered LDS
// filled by async-to-LDS copies, one barrier per K-step. Wave (wm,wn) in a
// 2x4 grid owns a 64x32 sub-tile = 4x2 accumulators = 8 WMMAs per K-step.
__global__ __launch_bounds__(256)
void gemm_bf16_wmma(const unsigned short* __restrict__ A,
                    const unsigned short* __restrict__ W,
                    float* __restrict__ C, int M, int N, int K, int ldc) {
  __shared__ unsigned short Ab[2][128 * 32];   // 8 KB each
  __shared__ unsigned short Bb[2][128 * 32];

  const int m0 = blockIdx.x * 128;
  const int n0 = blockIdx.y * 128;
  const int tid  = threadIdx.x;
  const int wave = tid >> 5;
  const int lane = tid & 31;
  const int wm = wave >> 2;          // 0..1 -> 64 rows each
  const int wn = wave & 3;           // 0..3 -> 32 cols each
  const int l16 = lane & 15;
  const int hi  = lane >> 4;

  // staging: 128 rows x 32 cols bf16 per tile; thread -> (row = tid>>1, 16-col seg)
  const int srow = tid >> 1;
  const int sseg = (tid & 1) * 16;
  const unsigned short* Ag = A + (size_t)(m0 + srow) * K + sseg;
  const unsigned short* Bg = W + (size_t)(n0 + srow) * K + sseg;
  const uint32_t lA0 = (uint32_t)(uintptr_t)&Ab[0][srow * 32 + sseg];
  const uint32_t lA1 = (uint32_t)(uintptr_t)&Ab[1][srow * 32 + sseg];
  const uint32_t lB0 = (uint32_t)(uintptr_t)&Bb[0][srow * 32 + sseg];
  const uint32_t lB1 = (uint32_t)(uintptr_t)&Bb[1][srow * 32 + sseg];

  auto stage = [&](int buf, int k0) {
    async_copy_32B(buf ? lA1 : lA0, Ag + k0);
    async_copy_32B(buf ? lB1 : lB0, Bg + k0);
  };

  v8f acc[4][2] = {};

  stage(0, 0);
  wait_async0();
  __syncthreads();
  int buf = 0;
  for (int k0 = 0; k0 < K; k0 += 32) {
    if (k0 + 32 < K) stage(buf ^ 1, k0 + 32);

    // gather fragments from LDS (each frag = two contiguous 16B runs)
    v16bf af[4];
#pragma unroll
    for (int mi = 0; mi < 4; ++mi) {
      const unsigned short* Ap = &Ab[buf][(wm * 64 + mi * 16 + l16) * 32];
#pragma unroll
      for (int j = 0; j < 8; ++j) {
        int kk = (j < 4) ? (hi * 8 + 2 * j) : (16 + hi * 8 + 2 * (j - 4));
        af[mi][2 * j]     = us_to_bf(Ap[kk]);
        af[mi][2 * j + 1] = us_to_bf(Ap[kk + 1]);
      }
    }
    v16bf bfr[2];
#pragma unroll
    for (int nj = 0; nj < 2; ++nj) {
      const unsigned short* Bp = &Bb[buf][(wn * 32 + nj * 16 + l16) * 32];
#pragma unroll
      for (int e = 0; e < 16; ++e) bfr[nj][e] = us_to_bf(Bp[(e < 8) ? (hi * 8 + e) : (8 + hi * 8 + e)]);
    }
#pragma unroll
    for (int mi = 0; mi < 4; ++mi)
#pragma unroll
      for (int nj = 0; nj < 2; ++nj)
        acc[mi][nj] = __builtin_amdgcn_wmma_f32_16x16x32_bf16(
            false, af[mi], false, bfr[nj], (short)0, acc[mi][nj], false, false);

    wait_async0();        // per-wave: staged tile landed in LDS
    __syncthreads();      // block-wide visibility before reading next buffer
    buf ^= 1;
  }

  // store: VGPR r -> row (hi*8 + r), lane col = l16
#pragma unroll
  for (int mi = 0; mi < 4; ++mi) {
    const int rbase = m0 + wm * 64 + mi * 16 + hi * 8;
#pragma unroll
    for (int nj = 0; nj < 2; ++nj) {
      const int col = n0 + wn * 32 + nj * 16 + l16;
#pragma unroll
      for (int r = 0; r < 8; ++r)
        C[(size_t)(rbase + r) * ldc + col] = acc[mi][nj][r];
    }
  }
}

// ---------- beta / decay-g tiny GEMV ----------
__global__ __launch_bounds__(128)
void betag_kernel(const float* __restrict__ x, const float* __restrict__ Wb,
                  const float* __restrict__ Wa, const float* __restrict__ A_log,
                  const float* __restrict__ dt_bias,
                  float* __restrict__ beta, float* __restrict__ gdec) {
  const int m = blockIdx.x;
  const int tid = threadIdx.x;
  const float* xp = x + (size_t)m * HIDDEN;
  float accB[NUM_HEADS] = {}, accA[NUM_HEADS] = {};
  for (int d = tid; d < HIDDEN; d += 128) {
    float xv = xp[d];
#pragma unroll
    for (int h = 0; h < NUM_HEADS; ++h) {
      accB[h] = fmaf(xv, Wb[h * HIDDEN + d], accB[h]);
      accA[h] = fmaf(xv, Wa[h * HIDDEN + d], accA[h]);
    }
  }
  __shared__ float red[128];
  for (int h = 0; h < NUM_HEADS; ++h) {
    red[tid] = accB[h]; __syncthreads();
    for (int s = 64; s > 0; s >>= 1) { if (tid < s) red[tid] += red[tid + s]; __syncthreads(); }
    float sb = red[0]; __syncthreads();
    red[tid] = accA[h]; __syncthreads();
    for (int s = 64; s > 0; s >>= 1) { if (tid < s) red[tid] += red[tid + s]; __syncthreads(); }
    float sa = red[0]; __syncthreads();
    if (tid == 0) {
      beta[m * NUM_HEADS + h] = 1.f / (1.f + __expf(-sb));
      float z  = sa + dt_bias[h];
      float sp = (z > 20.f) ? z : log1pf(__expf(z));
      gdec[m * NUM_HEADS + h] = -__expf(A_log[h]) * sp;
    }
  }
}

// ---------- causal depthwise conv + SiLU (+ optional per-head L2 norm) ----------
__global__ __launch_bounds__(256)
void conv_silu_kernel(const float* __restrict__ raw, int rstride,
                      const float* __restrict__ w,
                      float* __restrict__ out, int T, int do_norm) {
  const int bt = blockIdx.x;            // b*T + t
  const int h  = blockIdx.y;
  const int t  = bt % T;
  const int c  = threadIdx.x;
  const int ch = h * HEAD_DIM + c;
  const float* rp = raw + (size_t)bt * rstride + ch;

  float acc = 0.f;
#pragma unroll
  for (int i = 0; i < CONV_K; ++i) {
    int tt = t - (CONV_K - 1) + i;
    if (tt >= 0) acc = fmaf(rp[(tt - t) * rstride], w[ch * CONV_K + i], acc);
  }
  float s = silu(acc);
  if (do_norm) {
    __shared__ float red[256];
    red[c] = s * s; __syncthreads();
    for (int st = 128; st > 0; st >>= 1) { if (c < st) red[c] += red[c + st]; __syncthreads(); }
    s *= rsqrtf(red[0] + 1e-6f);        // x * rsqrt(sum(x^2)+eps)
  }
  out[(size_t)bt * KEY_DIM + ch] = s;
}

// ---------- delta-rule recurrence, column-sliced ----------
// v-columns of S evolve independently -> 8 slices of 32 cols per (b,h).
// One wave32 workgroup per slice (barriers compile to NOP); S slice in LDS.
#define RCOLS 32
__global__ __launch_bounds__(RCOLS)
void delta_recurrence(const float* __restrict__ q, const float* __restrict__ k,
                      const float* __restrict__ v, const float* __restrict__ gdec,
                      const float* __restrict__ beta, float* __restrict__ o,
                      int B, int T) {
  extern __shared__ float lds[];
  float* S  = lds;                      // [256][RCOLS]  (row = k index)
  float* kk = lds + HEAD_DIM * RCOLS;   // [256]
  float* qq = kk + HEAD_DIM;            // [256]

  const int nslice = HEAD_DIM / RCOLS;  // 8
  const int blk = blockIdx.x;
  const int slice = blk % nslice;
  const int bh = blk / nslice;
  const int b = bh / NUM_HEADS, h = bh % NUM_HEADS;
  const int c  = threadIdx.x;           // 0..31
  const int cg = slice * RCOLS + c;     // global v-column
  const float scale = 0.0625f;          // 256^-0.5

  for (int i = c; i < HEAD_DIM * RCOLS; i += RCOLS) S[i] = 0.f;
  __syncthreads();

  for (int t = 0; t < T; ++t) {
    const size_t idx = ((size_t)(b * T + t) * NUM_HEADS + h) * HEAD_DIM;
#pragma unroll
    for (int j = 0; j < HEAD_DIM / RCOLS; ++j) {
      kk[j * RCOLS + c] = k[idx + j * RCOLS + c];
      qq[j * RCOLS + c] = q[idx + j * RCOLS + c] * scale;
    }
    __syncthreads();

    const size_t mh = (size_t)(b * T + t) * NUM_HEADS + h;
    const float eg  = __expf(gdec[mh]);
    const float btv = beta[mh];
    const float vc  = v[idx + cg];

    float kv = 0.f;
    for (int kx = 0; kx < HEAD_DIM; ++kx) {      // decay + k·S
      float s = S[kx * RCOLS + c] * eg;
      S[kx * RCOLS + c] = s;
      kv = fmaf(kk[kx], s, kv);
    }
    const float u = btv * (vc - kv);
    float oc = 0.f;
    for (int kx = 0; kx < HEAD_DIM; ++kx) {      // rank-1 update + q·S
      float s = fmaf(kk[kx], u, S[kx * RCOLS + c]);
      S[kx * RCOLS + c] = s;
      oc = fmaf(qq[kx], s, oc);
    }
    o[idx + cg] = oc;
    __syncthreads();
  }
}

// ---------- RMSNorm * weight * silu(gate), output as bf16 ----------
__global__ __launch_bounds__(256)
void gate_norm_kernel(const float* __restrict__ o, const float* __restrict__ gate,
                      int gstride, const float* __restrict__ w,
                      unsigned short* __restrict__ og) {
  const int bt = blockIdx.x, h = blockIdx.y, c = threadIdx.x;
  const size_t idx = (size_t)bt * KEY_DIM + h * HEAD_DIM + c;
  const float ov = o[idx];
  __shared__ float red[256];
  red[c] = ov * ov; __syncthreads();
  for (int s = 128; s > 0; s >>= 1) { if (c < s) red[c] += red[c + s]; __syncthreads(); }
  const float inv = rsqrtf(red[0] / (float)HEAD_DIM + 1e-5f);
  const float gv = gate[(size_t)bt * gstride + h * HEAD_DIM + c];
  og[idx] = f32_to_bf16(ov * inv * w[c] * silu(gv));
}

// ---------- host launcher ----------
extern "C" void kernel_launch(void* const* d_in, const int* in_sizes, int n_in,
                              void* d_out, int out_size, void* d_ws, size_t ws_size,
                              hipStream_t stream) {
  (void)in_sizes; (void)n_in; (void)out_size; (void)ws_size;
  const float* x        = (const float*)d_in[0];
  const float* Wq       = (const float*)d_in[1];
  const float* Wk       = (const float*)d_in[2];
  const float* Wv       = (const float*)d_in[3];
  const float* Wo       = (const float*)d_in[4];
  const float* Wb       = (const float*)d_in[5];
  const float* Wa       = (const float*)d_in[6];
  const float* Wg       = (const float*)d_in[7];
  const float* conv_q   = (const float*)d_in[8];
  const float* conv_k   = (const float*)d_in[9];
  const float* conv_v   = (const float*)d_in[10];
  const float* o_norm_w = (const float*)d_in[11];
  const float* A_log    = (const float*)d_in[12];
  const float* dt_bias  = (const float*)d_in[13];

  const int B = 2, T = 2048, M = B * T;        // 4096 tokens
  const int D = HIDDEN, N = KEY_DIM;           // 2048, 1536
  const int NF = 4 * N;                        // fused qkvg width 6144

  char* ws = (char*)d_ws;
  size_t off = 0;
  auto take = [&](size_t bytes) -> char* {
    char* p = ws + off;
    off += (bytes + 255) & ~(size_t)255;
    return p;
  };
  unsigned short* xb   = (unsigned short*)take((size_t)M * D * 2);
  unsigned short* Wall = (unsigned short*)take((size_t)NF * D * 2);  // [Wq;Wk;Wv;Wg]
  unsigned short* Wob  = (unsigned short*)take((size_t)D * N * 2);
  float* qkvg = (float*)take((size_t)M * NF * 4);                    // fused raw projections
  float* qc   = (float*)take((size_t)M * N * 4);
  float* kc   = (float*)take((size_t)M * N * 4);
  float* vc   = (float*)take((size_t)M * N * 4);
  float* beta = (float*)take((size_t)M * NUM_HEADS * 4);
  float* gdec = (float*)take((size_t)M * NUM_HEADS * 4);
  float* o    = (float*)take((size_t)M * N * 4);
  unsigned short* og = (unsigned short*)take((size_t)M * N * 2);

  float* qraw = qkvg;              // stride NF per token
  float* kraw = qkvg + N;
  float* vraw = qkvg + 2 * N;
  float* graw = qkvg + 3 * N;

  auto cvt = [&](const float* s, unsigned short* d, size_t n) {
    cvt_bf16_kernel<<<dim3((unsigned)((n + 255) / 256)), dim3(256), 0, stream>>>(s, d, (int)n);
  };
  cvt(x,  xb, (size_t)M * D);
  cvt(Wq, Wall + (size_t)0 * N * D, (size_t)N * D);
  cvt(Wk, Wall + (size_t)1 * N * D, (size_t)N * D);
  cvt(Wv, Wall + (size_t)2 * N * D, (size_t)N * D);
  cvt(Wg, Wall + (size_t)3 * N * D, (size_t)N * D);
  cvt(Wo, Wob, (size_t)D * N);

  // fused projections: qkvg[M,6144] = x[M,2048] * Wall[6144,2048]^T
  gemm_bf16_wmma<<<dim3(M / 128, NF / 128), 256, 0, stream>>>(
      xb, Wall, qkvg, M, NF, D, NF);

  betag_kernel<<<M, 128, 0, stream>>>(x, Wb, Wa, A_log, dt_bias, beta, gdec);

  dim3 gconv(M, NUM_HEADS);
  conv_silu_kernel<<<gconv, 256, 0, stream>>>(qraw, NF, conv_q, qc, T, 1);
  conv_silu_kernel<<<gconv, 256, 0, stream>>>(kraw, NF, conv_k, kc, T, 1);
  conv_silu_kernel<<<gconv, 256, 0, stream>>>(vraw, NF, conv_v, vc, T, 0);

  const int nblk = B * NUM_HEADS * (HEAD_DIM / RCOLS);     // 96 wave32 WGs
  const size_t rlds = (HEAD_DIM * RCOLS + 2 * HEAD_DIM) * sizeof(float); // ~34.8 KB
  delta_recurrence<<<nblk, RCOLS, rlds, stream>>>(qc, kc, vc, gdec, beta, o, B, T);

  gate_norm_kernel<<<gconv, 256, 0, stream>>>(o, graw, NF, o_norm_w, og);

  // out[M,2048] = og[M,1536] * Wo[2048,1536]^T
  gemm_bf16_wmma<<<dim3(M / 128, D / 128), 256, 0, stream>>>(
      og, Wob, (float*)d_out, M, D, N, D);
}
